// SwitchLayer_90228672954407
// MI455X (gfx1250) — matrix operand order repaired and checked by reading, not compiled, not run on hardware
//
#include <hip/hip_runtime.h>
#include <math.h>
#include <stdint.h>

// ---------------------------------------------------------------------------
// Dense MoE (Switch-style, dense combine) for MI455X / gfx1250.
//   B=4, S=2048, D=1024, F=4096, E=8, topk=2.  T = B*S = 8192 tokens.
// ~1.1 TFLOP vs ~0.34 GB -> compute-bound -> bf16 WMMA (16x16x32), f32 accum.
// Mainloop: double-buffered LDS fed by global_load_async_to_lds_b128 (ASYNCcnt).
// ---------------------------------------------------------------------------

#define T_TOK 8192
#define DDIM  1024
#define FDIM  4096
#define NEXP  8

#define USE_ASYNC 1

// Block tile 256(M) x 128(N), K-step 32. 8 waves = 4(M) x 2(N), wave tile 64x64.
#define BM 256
#define BN 128
#define BK 32

typedef __attribute__((ext_vector_type(16))) __bf16 v16bf;
typedef __attribute__((ext_vector_type(8)))  float  v8f;

union Frag {
    v16bf v;
    unsigned int u[8];
};

__device__ __forceinline__ unsigned short f32_to_bf16(float f) {
    unsigned int u = __float_as_uint(f);
    u += 0x7FFFu + ((u >> 16) & 1u);      // round-to-nearest-even
    return (unsigned short)(u >> 16);
}

// A fragment: 16x32 bf16, row M = lane&15.
// lanes 0-15: VGPR0-3 K=0..7, VGPR4-7 K=16..23 ; lanes16-31: K=8..15 / 24..31
__device__ __forceinline__ void load_fragA(Frag& f, const unsigned short* rowp, int g) {
#pragma unroll
    for (int j = 0; j < 4; j++) {
        f.u[j]     = *(const unsigned int*)(rowp + g + 2 * j);
        f.u[4 + j] = *(const unsigned int*)(rowp + 16 + g + 2 * j);
    }
}

// B fragment: 32x16 bf16 stored N-major (stride 32) in LDS, col N = lane&15.
// lanes 0-15: VGPR0-3 K=0..7, VGPR4-7 K=16..23 ; lanes16-31: K=8..15 / 24..31
__device__ __forceinline__ void load_fragB(Frag& f, const unsigned short* rowp, int kb) {
#pragma unroll
    for (int v = 0; v < 8; v++)
        f.u[v] = *(const unsigned int*)(rowp + (v < 4 ? kb + 2 * v : 16 + kb + 2 * (v - 4)));
}

// Issue one K-stage of tile loads: A is BM x BK, B is BN x BK (both K-contig).
// 256 threads, 16B chunks: A = 4/thread, B = 2/thread (6 async ops per wave).
__device__ __forceinline__ void issue_tiles(const unsigned short* __restrict__ A,
                                            const unsigned short* __restrict__ Bt,
                                            int ldA, int ldB, int mbase, int nbase,
                                            int k0, int tid,
                                            unsigned short* lAbuf, unsigned short* lBbuf) {
#if USE_ASYNC
#pragma unroll
    for (int p = 0; p < 4; p++) {                 // A: BM*BK/8 = 1024 chunks
        const int ch  = tid + p * 256;
        const int row = ch >> 2;
        const int kc  = (ch & 3) * 8;
        unsigned dst = (unsigned)(uintptr_t)(lAbuf + row * BK + kc);
        unsigned long long src =
            (unsigned long long)(uintptr_t)(A + (size_t)(mbase + row) * ldA + k0 + kc);
        asm volatile("global_load_async_to_lds_b128 %0, %1, off"
                     :: "v"(dst), "v"(src) : "memory");
    }
#pragma unroll
    for (int p = 0; p < 2; p++) {                 // B: BN*BK/8 = 512 chunks
        const int ch  = tid + p * 256;
        const int row = ch >> 2;
        const int kc  = (ch & 3) * 8;
        unsigned dst = (unsigned)(uintptr_t)(lBbuf + row * BK + kc);
        unsigned long long src =
            (unsigned long long)(uintptr_t)(Bt + (size_t)(nbase + row) * ldB + k0 + kc);
        asm volatile("global_load_async_to_lds_b128 %0, %1, off"
                     :: "v"(dst), "v"(src) : "memory");
    }
    asm volatile("s_wait_xcnt 0x0" ::: "memory"); // addr VGPRs safe to reuse
#else
#pragma unroll
    for (int p = 0; p < 4; p++) {
        const int ch  = tid + p * 256;
        const int row = ch >> 2;
        const int kc  = (ch & 3) * 8;
        *(uint4*)(lAbuf + row * BK + kc) =
            *(const uint4*)(A + (size_t)(mbase + row) * ldA + k0 + kc);
    }
#pragma unroll
    for (int p = 0; p < 2; p++) {
        const int ch  = tid + p * 256;
        const int row = ch >> 2;
        const int kc  = (ch & 3) * 8;
        *(uint4*)(lBbuf + row * BK + kc) =
            *(const uint4*)(Bt + (size_t)(nbase + row) * ldB + k0 + kc);
    }
#endif
}

// One K-stage of WMMAs for a 64x64 wave tile (4x4 16x16 subtiles).
__device__ __forceinline__ void compute_stage(const unsigned short* la,
                                              const unsigned short* lb,
                                              int waveM, int waveN, int lane,
                                              v8f c[4][4]) {
    const int g  = (lane >> 4) * 8;
    const int kb = (lane >> 4) * 8;
    Frag a[4];
#pragma unroll
    for (int ms = 0; ms < 4; ms++)
        load_fragA(a[ms], la + (waveM * 64 + ms * 16 + (lane & 15)) * BK, g);
#pragma unroll
    for (int ns = 0; ns < 4; ns++) {
        Frag b;
        load_fragB(b, lb + (waveN * 64 + ns * 16 + (lane & 15)) * BK, kb);
#pragma unroll
        for (int ms = 0; ms < 4; ms++)
            c[ms][ns] = __builtin_amdgcn_wmma_f32_16x16x32_bf16(
                false, a[ms].v, false, b.v, (short)0, c[ms][ns], false, false);
    }
}

// ---------------------------------------------------------------------------
// Gate: one wave per token. logits -> softmax -> top-2 -> renormalize.
// ---------------------------------------------------------------------------
__global__ __launch_bounds__(256)
void gate_kernel(const float* __restrict__ x, const float* __restrict__ Wg,
                 float* __restrict__ P) {
    const int token = blockIdx.x * 8 + (threadIdx.x >> 5);
    const int lane  = threadIdx.x & 31;
    const float* xr = x + (size_t)token * DDIM;

    float acc[NEXP];
#pragma unroll
    for (int e = 0; e < NEXP; e++) acc[e] = 0.0f;

    for (int d = lane; d < DDIM; d += 32) {
        const float xv = xr[d];
        const float* wr = Wg + (size_t)d * NEXP;
#pragma unroll
        for (int e = 0; e < NEXP; e++) acc[e] = fmaf(xv, wr[e], acc[e]);
    }
#pragma unroll
    for (int off = 16; off > 0; off >>= 1) {
#pragma unroll
        for (int e = 0; e < NEXP; e++) acc[e] += __shfl_xor(acc[e], off);
    }
    float mx = acc[0];
#pragma unroll
    for (int e = 1; e < NEXP; e++) mx = fmaxf(mx, acc[e]);
    float p[NEXP], s = 0.0f;
#pragma unroll
    for (int e = 0; e < NEXP; e++) { p[e] = __expf(acc[e] - mx); s += p[e]; }
    const float inv = 1.0f / s;
#pragma unroll
    for (int e = 0; e < NEXP; e++) p[e] *= inv;
    int i1 = 0;
#pragma unroll
    for (int e = 1; e < NEXP; e++) if (p[e] > p[i1]) i1 = e;
    int i2 = (i1 == 0) ? 1 : 0;
#pragma unroll
    for (int e = 0; e < NEXP; e++) if (e != i1 && p[e] > p[i2]) i2 = e;
    const float denom = p[i1] + p[i2] + 1e-9f;
    if (lane < NEXP) {
        const float o = (lane == i1 || lane == i2) ? p[lane] / denom : 0.0f;
        P[(size_t)token * NEXP + lane] = o;
    }
}

// ---------------------------------------------------------------------------
// Elementwise f32 -> bf16 (vectorized x4)
// ---------------------------------------------------------------------------
__global__ __launch_bounds__(256)
void convert_f32_bf16(const float* __restrict__ src, unsigned short* __restrict__ dst,
                      int n4) {
    int i = blockIdx.x * blockDim.x + threadIdx.x;
    if (i >= n4) return;
    float4 v = ((const float4*)src)[i];
    union { unsigned short s[4]; uint2 u; } o;
    o.s[0] = f32_to_bf16(v.x); o.s[1] = f32_to_bf16(v.y);
    o.s[2] = f32_to_bf16(v.z); o.s[3] = f32_to_bf16(v.w);
    ((uint2*)dst)[i] = o.u;
}

// ---------------------------------------------------------------------------
// Tiled transpose + convert: src [E][rows][cols] f32 -> dst [E][cols][rows] bf16
// ---------------------------------------------------------------------------
__global__ __launch_bounds__(256)
void transpose_convert(const float* __restrict__ src, unsigned short* __restrict__ dst,
                       int rows, int cols) {
    __shared__ float tile[32][33];
    const size_t mat = (size_t)rows * cols;
    src += (size_t)blockIdx.z * mat;
    dst += (size_t)blockIdx.z * mat;
    const int c0 = blockIdx.x * 32, r0 = blockIdx.y * 32;
    const int tx = threadIdx.x & 31, ty = threadIdx.x >> 5;  // 32 x 8
#pragma unroll
    for (int i = 0; i < 32; i += 8)
        tile[ty + i][tx] = src[(size_t)(r0 + ty + i) * cols + (c0 + tx)];
    __syncthreads();
#pragma unroll
    for (int i = 0; i < 32; i += 8)
        dst[(size_t)(c0 + ty + i) * rows + (r0 + tx)] = f32_to_bf16(tile[tx][ty + i]);
}

// ---------------------------------------------------------------------------
// GEMM1: H[t][f] = bf16( P[t][e] * gelu_exact( xb[t,:] . W1T[f,:] + b1[f] ) )
//   A: xb [8192][1024] bf16 (K-contig)   B: W1T (per e) [4096][1024] bf16
// ---------------------------------------------------------------------------
__global__ __launch_bounds__(256)
void gemm1_kernel(const unsigned short* __restrict__ A,
                  const unsigned short* __restrict__ Bt,
                  const float* __restrict__ b1,
                  const float* __restrict__ P, int e,
                  unsigned short* __restrict__ H) {
    __shared__ __align__(16) unsigned short lA[2][BM * BK];
    __shared__ __align__(16) unsigned short lB[2][BN * BK];
    const int tid = threadIdx.x;
    const int lane = tid & 31, wid = tid >> 5;
    const int waveM = wid & 3, waveN = wid >> 2;
    const int mbase = blockIdx.x * BM;
    const int nbase = blockIdx.y * BN;
    const int KS = DDIM / BK;

    v8f c[4][4];
#pragma unroll
    for (int i = 0; i < 4; i++)
#pragma unroll
        for (int j = 0; j < 4; j++) c[i][j] = (v8f){0, 0, 0, 0, 0, 0, 0, 0};

    issue_tiles(A, Bt, DDIM, DDIM, mbase, nbase, 0, tid, lA[0], lB[0]);
    for (int ki = 0; ki < KS; ki++) {
        if (ki + 1 < KS)
            issue_tiles(A, Bt, DDIM, DDIM, mbase, nbase, (ki + 1) * BK, tid,
                        lA[(ki + 1) & 1], lB[(ki + 1) & 1]);
#if USE_ASYNC
        if (ki + 1 < KS) asm volatile("s_wait_asynccnt 0x6" ::: "memory");
        else             asm volatile("s_wait_asynccnt 0x0" ::: "memory");
#endif
        __syncthreads();
        compute_stage(lA[ki & 1], lB[ki & 1], waveM, waveN, lane, c);
        __syncthreads();
    }

    // epilogue: bias + exact gelu + gate scale, store bf16
#pragma unroll
    for (int ms = 0; ms < 4; ms++) {
        const int m0 = mbase + waveM * 64 + ms * 16 + (lane >> 4) * 8;
#pragma unroll
        for (int ns = 0; ns < 4; ns++) {
            const int n = nbase + waveN * 64 + ns * 16 + (lane & 15);
            const float bias = b1[n];
#pragma unroll
            for (int i = 0; i < 8; i++) {
                const int m = m0 + i;
                const float v  = c[ms][ns][i] + bias;
                const float gl = 0.5f * v * (1.0f + erff(v * 0.70710678118654752f));
                const float w  = P[(size_t)m * NEXP + e];
                H[(size_t)m * FDIM + n] = f32_to_bf16(gl * w);
            }
        }
    }
}

// ---------------------------------------------------------------------------
// GEMM2: Y[t][d] (+)= H[t,:] . W2T[d,:] + P[t][e]*b2[d]
//   A: H [8192][4096] bf16   B: W2T (per e) [1024][4096] bf16
// ---------------------------------------------------------------------------
__global__ __launch_bounds__(256)
void gemm2_kernel(const unsigned short* __restrict__ A,
                  const unsigned short* __restrict__ Bt,
                  const float* __restrict__ b2,
                  const float* __restrict__ P, int e, int beta,
                  float* __restrict__ Y) {
    __shared__ __align__(16) unsigned short lA[2][BM * BK];
    __shared__ __align__(16) unsigned short lB[2][BN * BK];
    const int tid = threadIdx.x;
    const int lane = tid & 31, wid = tid >> 5;
    const int waveM = wid & 3, waveN = wid >> 2;
    const int mbase = blockIdx.x * BM;
    const int nbase = blockIdx.y * BN;
    const int KS = FDIM / BK;

    v8f c[4][4];
#pragma unroll
    for (int i = 0; i < 4; i++)
#pragma unroll
        for (int j = 0; j < 4; j++) c[i][j] = (v8f){0, 0, 0, 0, 0, 0, 0, 0};

    issue_tiles(A, Bt, FDIM, FDIM, mbase, nbase, 0, tid, lA[0], lB[0]);
    for (int ki = 0; ki < KS; ki++) {
        if (ki + 1 < KS)
            issue_tiles(A, Bt, FDIM, FDIM, mbase, nbase, (ki + 1) * BK, tid,
                        lA[(ki + 1) & 1], lB[(ki + 1) & 1]);
#if USE_ASYNC
        if (ki + 1 < KS) asm volatile("s_wait_asynccnt 0x6" ::: "memory");
        else             asm volatile("s_wait_asynccnt 0x0" ::: "memory");
#endif
        __syncthreads();
        compute_stage(lA[ki & 1], lB[ki & 1], waveM, waveN, lane, c);
        __syncthreads();
    }

#pragma unroll
    for (int ms = 0; ms < 4; ms++) {
        const int m0 = mbase + waveM * 64 + ms * 16 + (lane >> 4) * 8;
#pragma unroll
        for (int ns = 0; ns < 4; ns++) {
            const int n = nbase + waveN * 64 + ns * 16 + (lane & 15);
            const float bias = b2[n];
#pragma unroll
            for (int i = 0; i < 8; i++) {
                const int m = m0 + i;
                const float w   = P[(size_t)m * NEXP + e];
                const size_t ix = (size_t)m * DDIM + n;
                const float val = c[ms][ns][i] + w * bias;
                Y[ix] = beta ? (Y[ix] + val) : val;
            }
        }
    }
}

// ---------------------------------------------------------------------------
extern "C" void kernel_launch(void* const* d_in, const int* in_sizes, int n_in,
                              void* d_out, int out_size, void* d_ws, size_t ws_size,
                              hipStream_t stream) {
    const float* x  = (const float*)d_in[0];
    const float* Wg = (const float*)d_in[1];
    const float* W1 = (const float*)d_in[2];
    const float* b1 = (const float*)d_in[3];
    const float* W2 = (const float*)d_in[4];
    const float* b2 = (const float*)d_in[5];
    float* y = (float*)d_out;

    char* ws = (char*)d_ws;
    size_t off = 0;
    float* P = (float*)(ws + off);                     off += (size_t)T_TOK * NEXP * 4;
    unsigned short* xb  = (unsigned short*)(ws + off); off += (size_t)T_TOK * DDIM * 2;
    unsigned short* W1T = (unsigned short*)(ws + off); off += (size_t)NEXP * DDIM * FDIM * 2;
    unsigned short* W2T = (unsigned short*)(ws + off); off += (size_t)NEXP * DDIM * FDIM * 2;
    unsigned short* Hs  = (unsigned short*)(ws + off); off += (size_t)T_TOK * FDIM * 2;

    // 1) gate probabilities (top-2 renormalized)
    gate_kernel<<<T_TOK / 8, 256, 0, stream>>>(x, Wg, P);

    // 2) precision staging: x -> bf16; W1 -> [E][F][D] bf16; W2 -> [E][D][F] bf16
    {
        const int n4 = T_TOK * DDIM / 4;
        convert_f32_bf16<<<(n4 + 255) / 256, 256, 0, stream>>>(x, xb, n4);
    }
    transpose_convert<<<dim3(FDIM / 32, DDIM / 32, NEXP), 256, 0, stream>>>(W1, W1T, DDIM, FDIM);
    transpose_convert<<<dim3(DDIM / 32, FDIM / 32, NEXP), 256, 0, stream>>>(W2, W2T, FDIM, DDIM);

    // 3) per-expert MLP: gemm1 (+gelu, gate-scaled) then gemm2 (accumulate into y)
    for (int e = 0; e < NEXP; e++) {
        gemm1_kernel<<<dim3(T_TOK / BM, FDIM / BN), 256, 0, stream>>>(
            xb, W1T + (size_t)e * DDIM * FDIM, b1 + (size_t)e * FDIM, P, e, Hs);
        gemm2_kernel<<<dim3(T_TOK / BM, DDIM / BN), 256, 0, stream>>>(
            Hs, W2T + (size_t)e * DDIM * FDIM, b2 + (size_t)e * DDIM, P, e, (e > 0), y);
    }
}
